// Stage0_12627203850254
// MI455X (gfx1250) — matrix operand order repaired and checked by reading, not compiled, not run on hardware
//
#include <hip/hip_runtime.h>

typedef __attribute__((ext_vector_type(16))) __bf16 v16bf;
typedef __attribute__((ext_vector_type(8)))  float  v8f;

#define BN_EPS 1e-5f

// ---------------- utility: zero a float buffer ----------------
__global__ void zero_f32(float* __restrict__ p, long long n) {
    long long i = (long long)blockIdx.x * blockDim.x + threadIdx.x;
    long long stride = (long long)gridDim.x * blockDim.x;
    for (; i < n; i += stride) p[i] = 0.0f;
}

// ---------------- conv1: 1 -> 64, scalar outer-product scatter ----------------
__global__ void conv1_scatter(const float* __restrict__ feats,
                              const float* __restrict__ W1,   // 27 x 1 x 64
                              const int* __restrict__ ki,     // 27 x M
                              const int* __restrict__ ko,     // 27 x M
                              float* __restrict__ out,        // (N+1) x 64, pre-zeroed
                              int M, int Nrows) {
    long long t = (long long)blockIdx.x * blockDim.x + threadIdx.x;
    long long total = (long long)27 * M * 4;
    if (t >= total) return;
    int quarter = (int)(t & 3);
    long long pair = t >> 2;
    int k = (int)(pair / M);
    int m = (int)(pair % M);
    int ir = ki[(long long)k * M + m];
    if (ir >= Nrows) return;                 // sentinel padding pair
    int orow = ko[(long long)k * M + m];
    float v = feats[ir];
    const float* w = W1 + k * 64 + quarter * 16;
    float* o = out + (long long)orow * 64 + quarter * 16;
#pragma unroll
    for (int c = 0; c < 16; ++c) atomicAdd(&o[c], v * w[c]);
}

// ---------------- BN statistics: per-channel sum & sumsq ----------------
template <int C>
__global__ void bn_stats(const float* __restrict__ x, int n, float* __restrict__ stats) {
    int gtid = blockIdx.x * blockDim.x + threadIdx.x;
    int c = gtid % C;
    int row0 = gtid / C;
    int rstride = (gridDim.x * blockDim.x) / C;   // blockDim multiple of C
    float s = 0.0f, s2 = 0.0f;
    for (int r = row0; r < n; r += rstride) {
        float v = x[(long long)r * C + c];
        s += v; s2 += v * v;
    }
    atomicAdd(&stats[c], s);
    atomicAdd(&stats[C + c], s2);
}

// ---------------- BN apply + ReLU (y may alias x) ----------------
template <int C>
__global__ void bn_apply(const float* __restrict__ x, float* __restrict__ y, int n,
                         const float* __restrict__ g, const float* __restrict__ be,
                         const float* __restrict__ stats) {
    long long i = (long long)blockIdx.x * blockDim.x + threadIdx.x;
    long long total = (long long)n * C;
    long long stride = (long long)gridDim.x * blockDim.x;
    float inv_n = 1.0f / (float)n;
    for (; i < total; i += stride) {
        int c = (int)(i % C);
        float mu = stats[c] * inv_n;
        float var = stats[C + c] * inv_n - mu * mu;
        float sc = g[c] * rsqrtf(var + BN_EPS);
        float v = (x[i] - mu) * sc + be[c];
        y[i] = v > 0.0f ? v : 0.0f;
    }
}

// ---------------- maxpool k2/s2: segment max (inputs >= 0 after ReLU) ----------------
__global__ void maxpool_seg(const float* __restrict__ x, const int* __restrict__ seg,
                            float* __restrict__ pooled, int n) {
    long long i = (long long)blockIdx.x * blockDim.x + threadIdx.x;
    long long total = (long long)n * 64;
    long long stride = (long long)gridDim.x * blockDim.x;
    for (; i < total; i += stride) {
        int r = (int)(i >> 6);
        int c = (int)(i & 63);
        float v = x[i];
        // post-ReLU values are non-negative: uint bit compare == float compare
        atomicMax((unsigned int*)&pooled[(long long)seg[r] * 64 + c], __float_as_uint(v));
    }
}

// ---------------- sparse conv via WMMA: 64 -> cout, gather-GEMM-scatter ----------------
// blockIdx.y = kernel offset k; blockIdx.z selects a 64-wide output-column slice.
// Each wave: B (W[k] slice, bf16) hoisted into registers once, then grid-strides over
// 16-row tiles of the (in,out) map: gather A rows (f32 -> bf16), 8x v_wmma
// 16x16x32 bf16 with f32 accumulate, scatter-add 16x64 via global f32 atomics.
__global__ __launch_bounds__(256) void sparse_conv_wmma(
    const float* __restrict__ x,      // (nin+1) x 64, row `nin` is the zero row
    const float* __restrict__ W,      // 27 x 64 x cout
    const int* __restrict__ ki,       // 27 x M
    const int* __restrict__ ko,       // 27 x M
    float* __restrict__ out,          // (nout+1) x cout, pre-zeroed (row nout = dummy)
    int M, int nin, int cout) {
    __shared__ __bf16 Wt[64 * 64];    // transposed [co_local][ci], bf16 (8 KB)

    const int k = blockIdx.y;
    const int col_base = blockIdx.z << 6;
    const float* Wk = W + (long long)k * 64 * cout + col_base;
    for (int i = threadIdx.x; i < 64 * 64; i += blockDim.x) {
        int ci = i >> 6, co = i & 63;
        Wt[co * 64 + ci] = (__bf16)Wk[(long long)ci * cout + co];
    }
    __syncthreads();

    const int lane = threadIdx.x & 31;
    const int wave = threadIdx.x >> 5;
    const int wpb  = blockDim.x >> 5;
    const int ml = lane & 15;   // row (A) / col (B,D) within tile
    const int hi = lane >> 4;   // K-half selector
    const int ntiles = (M + 15) >> 4;
    const int* kik = ki + (long long)k * M;
    const int* kok = ko + (long long)k * M;

    // ---- hoist B into registers: invariant across all tiles of this wave ----
    v16bf b[4][2];
#pragma unroll
    for (int nc = 0; nc < 4; ++nc) {
        const __bf16* wc = &Wt[(nc * 16 + ml) * 64];
#pragma unroll
        for (int j = 0; j < 8; ++j) {
            b[nc][0][2 * j]     = wc[8 * hi + 2 * j + 2 * (j & 4)];
            b[nc][0][2 * j + 1] = wc[8 * hi + 2 * j + 2 * (j & 4) + 1];
            b[nc][1][2 * j]     = wc[32 + 8 * hi + 2 * j + 2 * (j & 4)];
            b[nc][1][2 * j + 1] = wc[32 + 8 * hi + 2 * j + 2 * (j & 4) + 1];
        }
    }

    v8f zv;
#pragma unroll
    for (int e = 0; e < 8; ++e) zv[e] = 0.0f;

    for (int tile = blockIdx.x * wpb + wave; tile < ntiles; tile += (int)gridDim.x * wpb) {
        // ---- gather A rows, cvt f32 -> bf16 in the ISA lane layout ----
        int p = tile * 16 + ml;
        int in_row = (p < M) ? kik[p] : nin;   // sentinel -> zero row
        const float* xr = x + (long long)in_row * 64;
        v16bf a0, a1;
#pragma unroll
        for (int j = 0; j < 4; ++j) {
            int c0 = 8 * hi + 2 * j;
            a0[2 * j]           = (__bf16)xr[c0];
            a0[2 * j + 1]       = (__bf16)xr[c0 + 1];
            a0[2 * (j + 4)]     = (__bf16)xr[16 + c0];
            a0[2 * (j + 4) + 1] = (__bf16)xr[16 + c0 + 1];
            a1[2 * j]           = (__bf16)xr[32 + c0];
            a1[2 * j + 1]       = (__bf16)xr[32 + c0 + 1];
            a1[2 * (j + 4)]     = (__bf16)xr[48 + c0];
            a1[2 * (j + 4) + 1] = (__bf16)xr[48 + c0 + 1];
        }

        v8f acc[4];
#pragma unroll
        for (int nc = 0; nc < 4; ++nc) acc[nc] = zv;

        // ---- GEMM: 2 K-chunks x 4 N-chunks of 16x16x32 bf16 WMMA ----
#pragma unroll
        for (int nc = 0; nc < 4; ++nc) {
            acc[nc] = __builtin_amdgcn_wmma_f32_16x16x32_bf16(
                false, a0, false, b[nc][0], (short)0, acc[nc], false, false);
            acc[nc] = __builtin_amdgcn_wmma_f32_16x16x32_bf16(
                false, a1, false, b[nc][1], (short)0, acc[nc], false, false);
        }

        // ---- scatter-add: D VGPR r holds (row = r + 8*hi, col = ml) ----
#pragma unroll
        for (int r = 0; r < 8; ++r) {
            int q = tile * 16 + r + 8 * hi;
            if (q < M) {
                int orow = kok[q];                  // may be sentinel (dummy row)
                float* op = out + (long long)orow * cout + col_base;
#pragma unroll
                for (int nc = 0; nc < 4; ++nc)
                    atomicAdd(&op[nc * 16 + ml], acc[nc][r]);
            }
        }
    }
}

extern "C" void kernel_launch(void* const* d_in, const int* in_sizes, int n_in,
                              void* d_out, int out_size, void* d_ws, size_t ws_size,
                              hipStream_t stream) {
    const float* feats = (const float*)d_in[0];
    const float* W1    = (const float*)d_in[1];
    const float* g1    = (const float*)d_in[3];
    const float* be1   = (const float*)d_in[4];
    const float* W2    = (const float*)d_in[5];
    const float* g2    = (const float*)d_in[7];
    const float* be2   = (const float*)d_in[8];
    const float* W3    = (const float*)d_in[9];
    const float* g3    = (const float*)d_in[11];
    const float* be3   = (const float*)d_in[12];
    const int* km_in   = (const int*)d_in[13];
    const int* km_out  = (const int*)d_in[14];
    const int* pool_seg= (const int*)d_in[15];
    const int* km2_in  = (const int*)d_in[16];
    const int* km2_out = (const int*)d_in[17];

    const int N  = in_sizes[0];          // active voxels, level 1
    const int M  = in_sizes[13] / 27;    // pairs per offset, level 1
    const int M2 = in_sizes[16] / 27;    // pairs per offset, pooled level
    const int NP = out_size / 128;       // pooled voxels

    // workspace layout (f32)
    float* x1     = (float*)d_ws;                          // (N+1) x 64
    float* x2     = x1 + (long long)(N + 1) * 64;          // (N+1) x 64
    float* pooled = x2 + (long long)(N + 1) * 64;          // (NP+1) x 64
    float* x3     = pooled + (long long)(NP + 1) * 64;     // (NP+1) x 128
    float* stats  = x3 + (long long)(NP + 1) * 128;        // 3 x 256
    float* stats1 = stats;
    float* stats2 = stats + 256;
    float* stats3 = stats + 512;

    // zero all accumulators (deterministic per-call state)
    zero_f32<<<2048, 256, 0, stream>>>(x1, (long long)(N + 1) * 64);
    zero_f32<<<2048, 256, 0, stream>>>(x2, (long long)(N + 1) * 64);
    zero_f32<<<1024, 256, 0, stream>>>(pooled, (long long)(NP + 1) * 64);
    zero_f32<<<1024, 256, 0, stream>>>(x3, (long long)(NP + 1) * 128);
    zero_f32<<<1, 256, 0, stream>>>(stats, 768);

    // stage 1: conv 1->64, BN+ReLU (conv bias cancels under mean-subtracting BN)
    {
        long long total = (long long)27 * M * 4;
        int blocks = (int)((total + 255) / 256);
        conv1_scatter<<<blocks, 256, 0, stream>>>(feats, W1, km_in, km_out, x1, M, N);
    }
    bn_stats<64><<<512, 256, 0, stream>>>(x1, N, stats1);
    bn_apply<64><<<2048, 256, 0, stream>>>(x1, x1, N, g1, be1, stats1);

    // stage 2: conv 64->64 (WMMA), BN+ReLU
    {
        int ntiles = (M + 15) / 16;
        int gx = (ntiles + 7) / 8;            // 8 waves per block
        if (gx > 128) gx = 128;               // amortize W->LDS fill + B hoist over ~tiles
        sparse_conv_wmma<<<dim3(gx, 27, 1), 256, 0, stream>>>(
            x1, W2, km_in, km_out, x2, M, N, 64);
    }
    bn_stats<64><<<512, 256, 0, stream>>>(x2, N, stats2);
    bn_apply<64><<<2048, 256, 0, stream>>>(x2, x2, N, g2, be2, stats2);

    // maxpool k2/s2: segment max
    maxpool_seg<<<2048, 256, 0, stream>>>(x2, pool_seg, pooled, N);

    // stage 3: conv 64->128 (WMMA, two 64-col slices via blockIdx.z), BN+ReLU -> d_out
    {
        int ntiles = (M2 + 15) / 16;
        int gx = (ntiles + 7) / 8;
        if (gx > 128) gx = 128;
        sparse_conv_wmma<<<dim3(gx, 27, 2), 256, 0, stream>>>(
            pooled, W3, km2_in, km2_out, x3, M2, NP, 128);
    }
    bn_stats<128><<<512, 256, 0, stream>>>(x3, NP, stats3);
    bn_apply<128><<<2048, 256, 0, stream>>>(x3, (float*)d_out, NP, g3, be3, stats3);
}
